// MCGConv_75376676045227
// MI455X (gfx1250) — matrix operand order repaired and checked by reading, not compiled, not run on hardware
//
#include <hip/hip_runtime.h>
#include <math.h>

// Problem constants (match reference)
#define Bq 2
#define Nq 8192
#define Sq 1024
#define NSq 32
#define Vq 8
#define CFq 64
#define CINq 70
#define COUTq 128
#define H1q 32
#define RADIUSq 0.2f
#define BN_EPSq 1e-5f
#define GSTR 96   // padded row stride for gf/gf2 (>= CINq, multiple of 32)

typedef __attribute__((ext_vector_type(16))) _Float16 v16h;
typedef __attribute__((ext_vector_type(8)))  float    v8f;

__device__ __forceinline__ float gelu_exact(float x) {
  // jax.nn.gelu(approximate=False) = x * 0.5 * (1 + erf(x/sqrt(2)))
  return 0.5f * x * (1.0f + erff(x * 0.70710678118654752440f));
}

__device__ __forceinline__ v8f wmma_f16(v16h a, v16h b, v8f c) {
  // D = A(16x32,f16) x B(32x16,f16) + C(16x16,f32)  -- codegen-confirmed builtin
  return __builtin_amdgcn_wmma_f32_16x16x32_f16(false, a, false, b, (short)0, c,
                                                false, false);
}

// A-matrix 16x32 f16 layout: element j of v16h -> k offset within 32-wide chunk.
// Per lane-half the 16 elements form one contiguous 16-element run (k = 16*half + j).
__device__ __forceinline__ int kmapA(int j, int half) {
  return j + 16 * half;
}

// ---------------------------------------------------------------------------
// Kernel 1: farthest point sampling. One block per batch, 256 threads,
// 1024 sequential steps with block-wide argmax (first-occurrence tie-break).
// ---------------------------------------------------------------------------
__global__ void k_fps(const float* __restrict__ xyz,
                      float* __restrict__ new_xyz_ws,
                      float* __restrict__ new_xyz_out) {
  const int b = blockIdx.x;
  const int t = threadIdx.x;  // 256 threads
  __shared__ float rv[256];
  __shared__ int   ri[256];
  __shared__ float sp[3];
  __shared__ int   s_last;
  const float* base = xyz + (size_t)b * Nq * 3;

  float dmin[Nq / 256];
#pragma unroll
  for (int i = 0; i < Nq / 256; ++i) dmin[i] = 3.4e38f;
  if (t == 0) s_last = 0;
  __syncthreads();

  for (int s = 0; s < Sq; ++s) {
    if (t == 0) {
      int l = s_last;
      sp[0] = base[l * 3 + 0];
      sp[1] = base[l * 3 + 1];
      sp[2] = base[l * 3 + 2];
    }
    __syncthreads();
    float px = sp[0], py = sp[1], pz = sp[2];
    if (t == 0) {
      float* o1 = new_xyz_ws + ((size_t)b * Sq + s) * 3;
      float* o2 = new_xyz_out + ((size_t)b * Sq + s) * 3;
      o1[0] = px; o1[1] = py; o1[2] = pz;
      o2[0] = px; o2[1] = py; o2[2] = pz;
    }
    float bv = -1.0f; int bj = 0;
#pragma unroll
    for (int i = 0; i < Nq / 256; ++i) {
      int j = i * 256 + t;
      float dx = base[j * 3 + 0] - px;
      float dy = base[j * 3 + 1] - py;
      float dz = base[j * 3 + 2] - pz;
      float d = dx * dx + dy * dy + dz * dz;
      float dm = fminf(dmin[i], d);
      dmin[i] = dm;
      if (dm > bv || (dm == bv && j < bj)) { bv = dm; bj = j; }
    }
    rv[t] = bv; ri[t] = bj;
    __syncthreads();
    for (int off = 128; off > 0; off >>= 1) {
      if (t < off) {
        float v2 = rv[t + off]; int i2 = ri[t + off];
        if (v2 > rv[t] || (v2 == rv[t] && i2 < ri[t])) { rv[t] = v2; ri[t] = i2; }
      }
      __syncthreads();
    }
    if (t == 0) s_last = ri[0];
    __syncthreads();
  }
}

// ---------------------------------------------------------------------------
// Kernel 2: gather mc = xyz[:, mc_idx, :]
// ---------------------------------------------------------------------------
__global__ void k_mc(const float* __restrict__ xyz, const int* __restrict__ mc_idx,
                     float* __restrict__ mc) {
  int t = threadIdx.x;
  if (t < Bq * Vq * 3) {
    int b = t / (Vq * 3);
    int r = t % (Vq * 3);
    int v = r / 3, d = r % 3;
    mc[t] = xyz[(size_t)b * Nq * 3 + (size_t)mc_idx[v] * 3 + d];
  }
}

// ---------------------------------------------------------------------------
// Kernel 3: weight preprocessing: f16, K-padded copies so GEMM B-fragments are
// unconditional contiguous loads.
// ---------------------------------------------------------------------------
__global__ void k_prepw(const float* __restrict__ W2_0, const float* __restrict__ Wr,
                        const float* __restrict__ W2_1,
                        _Float16* __restrict__ W20h, _Float16* __restrict__ Wrh,
                        _Float16* __restrict__ W21h) {
  int i = blockIdx.x * 256 + threadIdx.x;
  if (i < COUTq * GSTR) {
    int o = i / GSTR, c = i % GSTR;
    W20h[i] = (_Float16)(c < CINq ? W2_0[o * CINq + c] : 0.0f);
    Wrh[i]  = (_Float16)(c < CINq ? Wr[o * CINq + c] : 0.0f);
  }
  if (i < COUTq * COUTq) W21h[i] = (_Float16)W2_1[i];
}

// ---------------------------------------------------------------------------
// Kernel 4: ball query (first NS in index order) + grouping into gf
// (row stride GSTR, pad columns zeroed). One wave32 per (b,s).
// ---------------------------------------------------------------------------
__global__ void k_group(const float* __restrict__ xyz, const float* __restrict__ feat,
                        const float* __restrict__ new_xyz, int* __restrict__ gidx,
                        float* __restrict__ gf) {
  int wid = (int)((blockIdx.x * blockDim.x + threadIdx.x) >> 5);
  int lane = threadIdx.x & 31;
  if (wid >= Bq * Sq) return;
  int b = wid / Sq, s = wid % Sq;
  const float* cb = new_xyz + ((size_t)b * Sq + s) * 3;
  float cx = cb[0], cy = cb[1], cz = cb[2];
  const float* xb = xyz + (size_t)b * Nq * 3;
  const float* fb = feat + (size_t)b * Nq * CFq;
  float* gfb = gf + ((size_t)b * Sq + s) * NSq * GSTR;
  int* gib = gidx + ((size_t)b * Sq + s) * NSq;

  int count = 0;
  for (int chunk = 0; chunk < Nq && count < NSq; chunk += 32) {
    int j = chunk + lane;
    float x = xb[j * 3 + 0], y = xb[j * 3 + 1], z = xb[j * 3 + 2];
    float dx = x - cx, dy = y - cy, dz = z - cz;
    float d2 = dx * dx + dy * dy + dz * dz;
    bool within = d2 < RADIUSq * RADIUSq;
    unsigned long long m = __ballot(within);
    int prefix = __popcll(m & ((1ull << lane) - 1ull));
    int slot = count + prefix;
    if (within && slot < NSq) {
      gib[slot] = j;
      float* row = gfb + slot * GSTR;
      row[0] = x; row[1] = y; row[2] = z;
      row[3] = dx; row[4] = dy; row[5] = dz;
      const float* fr = fb + (size_t)j * CFq;
      for (int c = 0; c < CFq; ++c) row[6 + c] = fr[c];
      for (int c = CINq; c < GSTR; ++c) row[c] = 0.0f;  // pad
    }
    count += (int)__popcll(m);
  }
  int valid = count < NSq ? count : NSq;
  if (lane >= valid) {
    gib[lane] = -1;
    float* row = gfb + lane * GSTR;
    for (int c = 0; c < GSTR; ++c) row[c] = 0.0f;
  }
}

// ---------------------------------------------------------------------------
// Kernel 5: BN1 stats pass. One thread per (b,s,n,v); computes pre-activation
// a = W1_0 @ rel + b1_0 and accumulates per-channel sum / sumsq.
// ---------------------------------------------------------------------------
__global__ void k_bn1_stats(const float* __restrict__ gf, const float* __restrict__ mc,
                            const float* __restrict__ W1_0, const float* __restrict__ b1_0,
                            float* __restrict__ acc) {
  __shared__ float sacc[2 * H1q];
  int t = threadIdx.x;
  if (t < 2 * H1q) sacc[t] = 0.0f;
  __syncthreads();

  long e = (long)blockIdx.x * blockDim.x + t;  // < B*S*NS*V
  int v = (int)(e & (Vq - 1));
  long bsn = e >> 3;
  const float* row = gf + bsn * GSTR;
  float gx = row[0], gy = row[1], gz = row[2];
  int b = (int)(bsn >> 15);  // S*NS = 32768
  const float* mcv = mc + (b * Vq + v) * 3;
  float mx = mcv[0], my = mcv[1], mz = mcv[2];
  float ax = gx - mx, ay = gy - my, az = gz - mz;
  float eu = sqrtf(ax * ax + ay * ay + az * az);
  float rel[13] = {-ax, -ay, -az, ax, ay, az, eu, gx, gy, gz, mx, my, mz};

  int lane = t & 31;
  for (int o = 0; o < H1q; ++o) {
    float a = b1_0[o];
#pragma unroll
    for (int i = 0; i < 13; ++i) a += W1_0[o * 13 + i] * rel[i];
    float s1 = a, s2 = a * a;
#pragma unroll
    for (int off = 16; off > 0; off >>= 1) {
      s1 += __shfl_xor(s1, off);
      s2 += __shfl_xor(s2, off);
    }
    if (lane == 0) { atomicAdd(&sacc[o], s1); atomicAdd(&sacc[H1q + o], s2); }
  }
  __syncthreads();
  if (t < 2 * H1q) atomicAdd(&acc[t], sacc[t]);
}

// ---------------------------------------------------------------------------
// Finalize BN stats: mean + rsqrt(var+eps) per channel.
// ---------------------------------------------------------------------------
__global__ void k_bn_final(const float* __restrict__ acc, float* __restrict__ bn,
                           int C, float invM) {
  int c = threadIdx.x;
  if (c < C) {
    float mean = acc[c] * invM;
    float var = acc[C + c] * invM - mean * mean;
    bn[c] = mean;
    bn[C + c] = rsqrtf(var + BN_EPSq);
  }
}

// ---------------------------------------------------------------------------
// Kernel 6: fused MLP1 (13->32 bn gelu -> 70) + gumbel softmax agg -> gf2.
// One wave per (b,s,n): stage1 lanes=hidden channel, stage2 lanes=out channel.
// Writes gf2 with GSTR stride incl. zeroed pad columns.
// ---------------------------------------------------------------------------
__global__ void k_mlp1(const float* __restrict__ gf, const float* __restrict__ mc,
                       const int* __restrict__ gidx, const float* __restrict__ gumbel_u,
                       const float* __restrict__ W1_0, const float* __restrict__ b1_0,
                       const float* __restrict__ g1_0, const float* __restrict__ be1_0,
                       const float* __restrict__ W1_1, const float* __restrict__ b1_1,
                       const float* __restrict__ bn1, float* __restrict__ gf2) {
  __shared__ float sA[8][Vq][H1q];
  int t = threadIdx.x;
  int w = t >> 5, lane = t & 31;
  long bsn = (long)blockIdx.x * 8 + w;  // (b*S + s)*NS + n
  const float* row = gf + bsn * GSTR;
  float gx = row[0], gy = row[1], gz = row[2];
  int b = (int)(bsn >> 15);
  bool masked = gidx[bsn] < 0;

  {  // stage 1: hidden channel h = lane
    int h = lane;
    float m1 = bn1[h], i1 = bn1[H1q + h];
    float gg = g1_0[h], bb = be1_0[h];
    float wv[13];
#pragma unroll
    for (int i = 0; i < 13; ++i) wv[i] = W1_0[h * 13 + i];
    float bias = b1_0[h];
    for (int v = 0; v < Vq; ++v) {
      const float* mcv = mc + (b * Vq + v) * 3;
      float ax = gx - mcv[0], ay = gy - mcv[1], az = gz - mcv[2];
      float eu = sqrtf(ax * ax + ay * ay + az * az);
      float a = bias
        + wv[0] * (-ax) + wv[1] * (-ay) + wv[2] * (-az)
        + wv[3] * ax + wv[4] * ay + wv[5] * az
        + wv[6] * eu
        + wv[7] * gx + wv[8] * gy + wv[9] * gz
        + wv[10] * mcv[0] + wv[11] * mcv[1] + wv[12] * mcv[2];
      a = (a - m1) * i1 * gg + bb;
      sA[w][v][h] = gelu_exact(a);
    }
  }
  __syncthreads();

  const float* ub = gumbel_u + bsn * (long)(CINq * Vq);
  float* g2 = gf2 + bsn * GSTR;
  for (int ci = 0; ci < 3; ++ci) {
    int c = lane + 32 * ci;
    if (c < CINq) {
      float gfc = row[c];
      float p[Vq];
      for (int v = 0; v < Vq; ++v) {
        float a2 = b1_1[c];
#pragma unroll
        for (int h = 0; h < H1q; ++h) a2 += sA[w][v][h] * W1_1[c * H1q + h];
        p[v] = gfc * a2;
      }
      float z[Vq]; float zm = -3.4e38f;
#pragma unroll
      for (int v = 0; v < Vq; ++v) {
        float u = ub[c * Vq + v];
        float gum = -logf(-logf(u));
        z[v] = p[v] + gum;
        zm = fmaxf(zm, z[v]);
      }
      float den = 0.0f;
#pragma unroll
      for (int v = 0; v < Vq; ++v) { z[v] = expf(z[v] - zm); den += z[v]; }
      float agg = 0.0f;
#pragma unroll
      for (int v = 0; v < Vq; ++v) agg += (z[v] / den) * p[v];
      g2[c] = masked ? 0.0f : (agg + gfc);
    } else {
      g2[c] = 0.0f;  // pad columns
    }
  }
}

// ---------------------------------------------------------------------------
// Kernel 7: BN2 stats. One block per (b,s); WMMA GEMM w1 = gf2 @ W2_0^T + b2_0
// then per-channel sum/sumsq accumulation.
// ---------------------------------------------------------------------------
__global__ void k_bn2_stats(const float* __restrict__ gf2,
                            const _Float16* __restrict__ W20h,
                            const float* __restrict__ b2_0,
                            float* __restrict__ acc) {
  __shared__ _Float16 sA2[NSq][GSTR];
  int t = threadIdx.x;
  long bs = blockIdx.x;
  const float* gbase = gf2 + bs * (long)(NSq * GSTR);
  {
    _Float16* p = &sA2[0][0];
    for (int i = t; i < NSq * GSTR / 4; i += 256) {
      float4 v = ((const float4*)gbase)[i];
      p[4 * i + 0] = (_Float16)v.x;
      p[4 * i + 1] = (_Float16)v.y;
      p[4 * i + 2] = (_Float16)v.z;
      p[4 * i + 3] = (_Float16)v.w;
    }
  }
  __syncthreads();

  int w = t >> 5, lane = t & 31;
  int half = lane >> 4, lrow = lane & 15;
  int o = w * 16 + lrow;
  v8f c0 = {}, c1 = {};
#pragma unroll
  for (int kt = 0; kt < 3; ++kt) {
    int kb = kt * 32;
    v16h a0, a1, bf;
    const _Float16* wrow = W20h + o * GSTR + kb + 16 * half;
#pragma unroll
    for (int j = 0; j < 16; ++j) {
      int ka = kb + kmapA(j, half);
      a0[j] = sA2[lrow][ka];
      a1[j] = sA2[16 + lrow][ka];
      bf[j] = wrow[j];
    }
    c0 = wmma_f16(a0, bf, c0);
    c1 = wmma_f16(a1, bf, c1);
  }
  float bias = b2_0[o];
  float s1 = 0.0f, s2 = 0.0f;
#pragma unroll
  for (int r = 0; r < 8; ++r) {
    float x = c0[r] + bias; s1 += x; s2 += x * x;
    x = c1[r] + bias; s1 += x; s2 += x * x;
  }
  s1 += __shfl_xor(s1, 16);
  s2 += __shfl_xor(s2, 16);
  if (half == 0) {
    atomicAdd(&acc[o], s1);
    atomicAdd(&acc[COUTq + o], s2);
  }
}

// ---------------------------------------------------------------------------
// Kernel 8: fused MLP2. One block per (b,s); three WMMA GEMMs
// (gf2@W2_0^T -> bn gelu -> @W2_1^T, residual gf@Wr^T), gelu, max+sum over NS.
// ---------------------------------------------------------------------------
__global__ void k_mlp2(const float* __restrict__ gf, const float* __restrict__ gf2,
                       const _Float16* __restrict__ W20h, const float* __restrict__ b2_0,
                       const float* __restrict__ g2_0, const float* __restrict__ be2_0,
                       const _Float16* __restrict__ W21h, const float* __restrict__ b2_1,
                       const _Float16* __restrict__ Wrh, const float* __restrict__ br,
                       const float* __restrict__ bn2, float* __restrict__ out) {
  __shared__ _Float16 sA2[NSq][GSTR];
  __shared__ _Float16 sAg[NSq][GSTR];
  __shared__ _Float16 sW1[NSq][COUTq];
  int t = threadIdx.x;
  long bs = blockIdx.x;
  const float* g2base = gf2 + bs * (long)(NSq * GSTR);
  const float* gbase = gf + bs * (long)(NSq * GSTR);
  {
    _Float16* p2 = &sA2[0][0];
    _Float16* pg = &sAg[0][0];
    for (int i = t; i < NSq * GSTR / 4; i += 256) {
      float4 v = ((const float4*)g2base)[i];
      p2[4 * i + 0] = (_Float16)v.x;
      p2[4 * i + 1] = (_Float16)v.y;
      p2[4 * i + 2] = (_Float16)v.z;
      p2[4 * i + 3] = (_Float16)v.w;
      float4 u = ((const float4*)gbase)[i];
      pg[4 * i + 0] = (_Float16)u.x;
      pg[4 * i + 1] = (_Float16)u.y;
      pg[4 * i + 2] = (_Float16)u.z;
      pg[4 * i + 3] = (_Float16)u.w;
    }
  }
  __syncthreads();

  int w = t >> 5, lane = t & 31;
  int half = lane >> 4, lrow = lane & 15;
  int o = w * 16 + lrow;  // output channel owned by this lane

  // GEMM1: w1 = gf2 @ W2_0^T  (M=32, N=16 per wave, K=70 padded to 96)
  v8f c0 = {}, c1 = {};
#pragma unroll
  for (int kt = 0; kt < 3; ++kt) {
    int kb = kt * 32;
    v16h a0, a1, bf;
    const _Float16* wrow = W20h + o * GSTR + kb + 16 * half;
#pragma unroll
    for (int j = 0; j < 16; ++j) {
      int ka = kb + kmapA(j, half);
      a0[j] = sA2[lrow][ka];
      a1[j] = sA2[16 + lrow][ka];
      bf[j] = wrow[j];
    }
    c0 = wmma_f16(a0, bf, c0);
    c1 = wmma_f16(a1, bf, c1);
  }
  {
    float bias = b2_0[o];
    float m = bn2[o], iv = bn2[COUTq + o];
    float gg = g2_0[o], bb = be2_0[o];
#pragma unroll
    for (int r = 0; r < 8; ++r) {
      float x = (c0[r] + bias - m) * iv * gg + bb;
      sW1[r + 8 * half][o] = (_Float16)gelu_exact(x);
      x = (c1[r] + bias - m) * iv * gg + bb;
      sW1[16 + r + 8 * half][o] = (_Float16)gelu_exact(x);
    }
  }
  __syncthreads();

  // GEMM2: w2 = gelu_w1 @ W2_1^T (K=128)
  v8f d0 = {}, d1 = {};
#pragma unroll
  for (int kt = 0; kt < 4; ++kt) {
    int kb = kt * 32;
    v16h a0, a1, bf;
    const _Float16* wrow = W21h + o * COUTq + kb + 16 * half;
#pragma unroll
    for (int j = 0; j < 16; ++j) {
      int ka = kb + kmapA(j, half);
      a0[j] = sW1[lrow][ka];
      a1[j] = sW1[16 + lrow][ka];
      bf[j] = wrow[j];
    }
    d0 = wmma_f16(a0, bf, d0);
    d1 = wmma_f16(a1, bf, d1);
  }
  // GEMMres: res = gf @ Wr^T (K=70 padded to 96)
  v8f e0 = {}, e1 = {};
#pragma unroll
  for (int kt = 0; kt < 3; ++kt) {
    int kb = kt * 32;
    v16h a0, a1, bf;
    const _Float16* wrow = Wrh + o * GSTR + kb + 16 * half;
#pragma unroll
    for (int j = 0; j < 16; ++j) {
      int ka = kb + kmapA(j, half);
      a0[j] = sAg[lrow][ka];
      a1[j] = sAg[16 + lrow][ka];
      bf[j] = wrow[j];
    }
    e0 = wmma_f16(a0, bf, e0);
    e1 = wmma_f16(a1, bf, e1);
  }

  float b21 = b2_1[o], brr = br[o];
  float mx = -3.4e38f, sm = 0.0f;
#pragma unroll
  for (int r = 0; r < 8; ++r) {
    float x = gelu_exact(d0[r] + b21 + e0[r] + brr);
    mx = fmaxf(mx, x); sm += x;
    x = gelu_exact(d1[r] + b21 + e1[r] + brr);
    mx = fmaxf(mx, x); sm += x;
  }
  float mx2 = __shfl_xor(mx, 16);
  float sm2 = __shfl_xor(sm, 16);
  mx = fmaxf(mx, mx2); sm += sm2;
  if (half == 0) out[bs * COUTq + o] = mx + sm;
}

// ---------------------------------------------------------------------------
extern "C" void kernel_launch(void* const* d_in, const int* in_sizes, int n_in,
                              void* d_out, int out_size, void* d_ws, size_t ws_size,
                              hipStream_t stream) {
  const float* xyz   = (const float*)d_in[0];
  const float* f     = (const float*)d_in[1];
  const int*   mci   = (const int*)d_in[2];
  const float* gum   = (const float*)d_in[3];
  const float* W1_0  = (const float*)d_in[4];
  const float* b1_0  = (const float*)d_in[5];
  const float* g1_0  = (const float*)d_in[6];
  const float* be1_0 = (const float*)d_in[7];
  const float* W1_1  = (const float*)d_in[8];
  const float* b1_1  = (const float*)d_in[9];
  const float* W2_0  = (const float*)d_in[10];
  const float* b2_0  = (const float*)d_in[11];
  const float* g2_0  = (const float*)d_in[12];
  const float* be2_0 = (const float*)d_in[13];
  const float* W2_1  = (const float*)d_in[14];
  const float* b2_1  = (const float*)d_in[15];
  const float* Wr    = (const float*)d_in[16];
  const float* br    = (const float*)d_in[17];

  float* out = (float*)d_out;
  float* new_xyz_out = out + (size_t)Bq * Sq * COUTq;

  float* ws = (float*)d_ws;
  size_t off = 0;
  float* nxyz = ws + off;              off += (size_t)Bq * Sq * 3;
  int*   gidx = (int*)(ws + off);      off += (size_t)Bq * Sq * NSq;
  float* gf   = ws + off;              off += (size_t)Bq * Sq * NSq * GSTR;
  float* gf2  = ws + off;              off += (size_t)Bq * Sq * NSq * GSTR;
  float* mc   = ws + off;              off += (size_t)Bq * Vq * 3 + 2;  // later buffers 8B-aligned only
  float* acc1 = ws + off;              off += 2 * H1q;
  float* bn1  = ws + off;              off += 2 * H1q;
  float* acc2 = ws + off;              off += 2 * COUTq;
  float* bn2  = ws + off;              off += 2 * COUTq;
  _Float16* W20h = (_Float16*)(ws + off);  off += (size_t)COUTq * GSTR / 2;
  _Float16* Wrh  = (_Float16*)(ws + off);  off += (size_t)COUTq * GSTR / 2;
  _Float16* W21h = (_Float16*)(ws + off);  off += (size_t)COUTq * COUTq / 2;
  (void)ws_size; (void)in_sizes; (void)n_in; (void)out_size;

  k_fps<<<Bq, 256, 0, stream>>>(xyz, nxyz, new_xyz_out);
  k_mc<<<1, 64, 0, stream>>>(xyz, mci, mc);
  k_prepw<<<(COUTq * COUTq) / 256, 256, 0, stream>>>(W2_0, Wr, W2_1, W20h, Wrh, W21h);
  k_group<<<(Bq * Sq) / 8, 256, 0, stream>>>(xyz, f, nxyz, gidx, gf);

  hipMemsetAsync(acc1, 0, 2 * H1q * sizeof(float), stream);
  k_bn1_stats<<<(Bq * Sq * NSq * Vq) / 256, 256, 0, stream>>>(gf, mc, W1_0, b1_0, acc1);
  k_bn_final<<<1, 64, 0, stream>>>(acc1, bn1, H1q, 1.0f / (float)(Bq * Sq * NSq * Vq));

  k_mlp1<<<(Bq * Sq * NSq) / 8, 256, 0, stream>>>(gf, mc, gidx, gum, W1_0, b1_0, g1_0,
                                                  be1_0, W1_1, b1_1, bn1, gf2);

  hipMemsetAsync(acc2, 0, 2 * COUTq * sizeof(float), stream);
  k_bn2_stats<<<Bq * Sq, 256, 0, stream>>>(gf2, W20h, b2_0, acc2);
  k_bn_final<<<1, 128, 0, stream>>>(acc2, bn2, COUTq, 1.0f / (float)(Bq * Sq * NSq));

  k_mlp2<<<Bq * Sq, 256, 0, stream>>>(gf, gf2, W20h, b2_0, g2_0, be2_0, W21h, b2_1,
                                      Wrh, br, bn2, out);
}
